// PointCloudMPE_3272765080109
// MI455X (gfx1250) — matrix operand order repaired and checked by reading, not compiled, not run on hardware
//
#include <hip/hip_runtime.h>
#include <math.h>

#define PCMPE_INLINE __attribute__((always_inline)) inline

// ---------------------------------------------------------------------------
// Types for CDNA5 WMMA (wave32):  D(16x16 f32) = A(16x32 bf16) * B(32x16 bf16) + C
// ---------------------------------------------------------------------------
typedef __attribute__((ext_vector_type(16))) __bf16 bf16x16;
typedef __attribute__((ext_vector_type(8)))  __bf16 bf16x8;
typedef __attribute__((ext_vector_type(8)))  float  f32x8;

// Problem constants (from reference setup_inputs)
#define BATCH   8
#define NPTS    4096
#define NFEAT   64
#define NHID    128
#define KNN     16
#define NPOINTS (BATCH * NPTS)   // 32768

// ---------------------------------------------------------------------------
// Workspace layout (bytes). All offsets 256B aligned.
// ---------------------------------------------------------------------------
#define WS_CENTER   0            //  8*3 f32
#define WS_INV      256          //  32768*6 f32 = 786432
#define WS_WTS      786688       //  packed bf16 weights
#define WP_INV_W1   0            //  kt=1, nt=8  ->  4096 bf16 =  8192B
#define WP_INV_W2   8192         //  kt=4, nt=8  -> 16384 bf16 = 32768B
#define WP_FEAT_W1  40960        //  kt=2, nt=8  ->  8192 bf16 = 16384B
#define WP_FEAT_W2  57344        //  kt=4, nt=8  -> 32768B
#define WP_SH_W1    90112        //  kt=8, nt=8  -> 65536B
#define WP_SH_W2    155648       //  kt=4, nt=8  -> 32768B
#define WP_G0       188416       //  kt=4, nt=1  ->  4096B

// ===========================================================================
// Kernel 0: per-batch centroid (coords mean over N)
// ===========================================================================
__global__ void pcmpe_center_kernel(const float* __restrict__ coords,
                                    float* __restrict__ center) {
  __shared__ float sx[256], sy[256], sz[256];
  const int b = blockIdx.x, t = threadIdx.x;
  const float* cb = coords + (size_t)b * NPTS * 3;
  float ax = 0.f, ay = 0.f, az = 0.f;
  for (int i = t; i < NPTS; i += 256) {
    ax += cb[i * 3 + 0]; ay += cb[i * 3 + 1]; az += cb[i * 3 + 2];
  }
  sx[t] = ax; sy[t] = ay; sz[t] = az;
  __syncthreads();
  for (int s = 128; s > 0; s >>= 1) {
    if (t < s) { sx[t] += sx[t + s]; sy[t] += sy[t + s]; sz[t] += sz[t + s]; }
    __syncthreads();
  }
  if (t == 0) {
    center[b * 3 + 0] = sx[0] * (1.0f / NPTS);
    center[b * 3 + 1] = sy[0] * (1.0f / NPTS);
    center[b * 3 + 2] = sz[0] * (1.0f / NPTS);
  }
}

// ===========================================================================
// Kernel 1: kNN + covariance + closed-form 3x3 eigh + normals + invariants.
// One thread per query point; candidates tiled through LDS.
// Writes invariants[6] (f32, ws) and output grades 1..4 (d_out cols 1..15).
// ===========================================================================
__global__ void pcmpe_geom_kernel(const float* __restrict__ coords,
                                  const float* __restrict__ center,
                                  float* __restrict__ inv_out,
                                  float* __restrict__ out) {
  __shared__ float sx[256], sy[256], sz[256];
  const int p = blockIdx.x * 256 + threadIdx.x;   // global point
  const int b = p >> 12;                          // batch
  const int n = p & (NPTS - 1);                   // point in batch
  const float* cb = coords + ((size_t)b * NPTS) * 3;

  const float qx = cb[n * 3 + 0], qy = cb[n * 3 + 1], qz = cb[n * 3 + 2];

  float  d[KNN];
  int    id[KNN];
#pragma unroll
  for (int s = 0; s < KNN; ++s) { d[s] = 3.4e38f; id[s] = 0; }

  for (int t = 0; t < NPTS / 256; ++t) {
    __syncthreads();
    const int j = t * 256 + threadIdx.x;
    sx[threadIdx.x] = cb[j * 3 + 0];
    sy[threadIdx.x] = cb[j * 3 + 1];
    sz[threadIdx.x] = cb[j * 3 + 2];
    if (t + 1 < NPTS / 256)   // gfx1250 global_prefetch_b8 of next tile
      __builtin_prefetch(&cb[(t + 1) * 256 * 3 + threadIdx.x * 3], 0, 1);
    __syncthreads();
    for (int jj = 0; jj < 256; ++jj) {
      const float dx = sx[jj] - qx, dy = sy[jj] - qy, dz = sz[jj] - qz;
      const float dd = dx * dx + dy * dy + dz * dz;
      const int cidx = t * 256 + jj;
      if (cidx != n && dd < d[KNN - 1]) {
        float cd = dd; int ci = cidx;
#pragma unroll
        for (int s = 0; s < KNN; ++s) {
          if (cd < d[s]) { float td = d[s]; int ti = id[s]; d[s] = cd; id[s] = ci; cd = td; ci = ti; }
        }
      }
    }
  }

  // covariance of relative neighbors + mean radius
  float c00 = 0.f, c01 = 0.f, c02 = 0.f, c11 = 0.f, c12 = 0.f, c22 = 0.f, rad = 0.f;
#pragma unroll
  for (int s = 0; s < KNN; ++s) {
    const float* nb = cb + id[s] * 3;
    const float rx = nb[0] - qx, ry = nb[1] - qy, rz = nb[2] - qz;
    c00 += rx * rx; c01 += rx * ry; c02 += rx * rz;
    c11 += ry * ry; c12 += ry * rz; c22 += rz * rz;
    rad += sqrtf(rx * rx + ry * ry + rz * rz);
  }
  const float invk = 1.0f / KNN;
  c00 *= invk; c01 *= invk; c02 *= invk; c11 *= invk; c12 *= invk; c22 *= invk;
  rad *= invk;

  // closed-form eigenvalues of symmetric 3x3 (ascending e0<=e1<=e2)
  const float q  = (c00 + c11 + c22) * (1.0f / 3.0f);
  const float p1 = c01 * c01 + c02 * c02 + c12 * c12;
  const float p2 = (c00 - q) * (c00 - q) + (c11 - q) * (c11 - q) + (c22 - q) * (c22 - q) + 2.0f * p1;
  const float pp = sqrtf(p2 * (1.0f / 6.0f));
  float e0, e1, e2;
  if (pp < 1e-12f) {
    e0 = e1 = e2 = q;
  } else {
    const float ip  = 1.0f / pp;
    const float b00 = (c00 - q) * ip, b11 = (c11 - q) * ip, b22 = (c22 - q) * ip;
    const float b01 = c01 * ip, b02 = c02 * ip, b12 = c12 * ip;
    float r = 0.5f * (b00 * (b11 * b22 - b12 * b12)
                    - b01 * (b01 * b22 - b12 * b02)
                    + b02 * (b01 * b12 - b11 * b02));
    r = fminf(1.0f, fmaxf(-1.0f, r));
    const float phi = acosf(r) * (1.0f / 3.0f);
    e2 = q + 2.0f * pp * cosf(phi);
    e0 = q + 2.0f * pp * cosf(phi + 2.0943951023931953f);
    e1 = 3.0f * q - e2 - e0;
  }

  // eigenvector of smallest eigenvalue: largest cross product of rows of (A - e0 I)
  const float m00 = c00 - e0, m11 = c11 - e0, m22 = c22 - e0;
  float v0x = c01 * c12 - m11 * c02,  v0y = c02 * c01 - m00 * c12,  v0z = m00 * m11 - c01 * c01; // r0 x r1
  float v1x = c01 * m22 - c12 * c02,  v1y = c02 * c02 - m00 * m22,  v1z = m00 * c12 - c01 * c02; // r0 x r2
  float v2x = m11 * m22 - c12 * c12,  v2y = c12 * c02 - c01 * m22,  v2z = c01 * c12 - m11 * c02; // r1 x r2
  const float l0 = v0x * v0x + v0y * v0y + v0z * v0z;
  const float l1 = v1x * v1x + v1y * v1y + v1z * v1z;
  const float l2 = v2x * v2x + v2y * v2y + v2z * v2z;
  float vx, vy, vz, ll;
  if (l0 >= l1 && l0 >= l2)      { vx = v0x; vy = v0y; vz = v0z; ll = l0; }
  else if (l1 >= l2)             { vx = v1x; vy = v1y; vz = v1z; ll = l1; }
  else                           { vx = v2x; vy = v2y; vz = v2z; ll = l2; }
  if (ll < 1e-20f) { vx = 0.f; vy = 0.f; vz = 1.f; }
  else { const float rn = rsqrtf(ll); vx *= rn; vy *= rn; vz *= rn; }

  // orientation vs. outward direction
  const float ox = qx - center[b * 3 + 0];
  const float oy = qy - center[b * 3 + 1];
  const float oz = qz - center[b * 3 + 2];
  const float dt = vx * ox + vy * oy + vz * oz;
  const float sg = (dt < 0.f) ? -1.0f : 1.0f;   // sign(0) -> +1 per reference
  vx *= sg; vy *= sg; vz *= sg;
  const float nn2 = fmaxf(sqrtf(vx * vx + vy * vy + vz * vz), 1e-6f);
  vx /= nn2; vy /= nn2; vz /= nn2;

  const float crad   = sqrtf(ox * ox + oy * oy + oz * oz);
  const float eigsum = fmaxf(e0 + e1 + e2, 1e-6f);

  float* iv = inv_out + (size_t)p * 6;
  iv[0] = e0; iv[1] = e1; iv[2] = e2; iv[3] = rad; iv[4] = crad; iv[5] = e2 / eigsum;

  float* o = out + (size_t)p * 16;
  o[1]  = vx;  o[2]  = vy;  o[3]  = -vz;
  o[4]  = -(qx * vx + qy * vy + qz * vz);
  o[5]  = vx;  o[6]  = vy;  o[7]  = vz;
  o[8]  = 0.f; o[9]  = 0.f; o[10] = 0.f;
  o[11] = qx;  o[12] = qy;  o[13] = qz;  o[14] = 1.0f;
  o[15] = 0.f;
}

// ===========================================================================
// Kernel 2: pack f32 weight matrix [Kreal x Ncols] into WMMA B-fragment
// bf16 layout: frag(kt,nt) lane(n+16h) element j  <-  W[kt*32 + 16*(j>>3) + 8h + (j&7)][nt*16+n]
// ===========================================================================
__global__ void pcmpe_pack_kernel(const float* __restrict__ W,
                                  __bf16* __restrict__ outp,
                                  int Kreal, int Ncols, int kTiles, int nTiles) {
  const int t = blockIdx.x * 256 + threadIdx.x;
  const int total = kTiles * nTiles * 32;
  if (t >= total) return;
  const int lane = t & 31;
  const int frag = t >> 5;
  const int nt = frag % nTiles;
  const int kt = frag / nTiles;
  const int h = lane >> 4, nn = lane & 15;
  const int col = nt * 16 + nn;
#pragma unroll
  for (int j = 0; j < 16; ++j) {
    const int kk = kt * 32 + 16 * (j >> 3) + 8 * h + (j & 7);
    const float v = (kk < Kreal && col < Ncols) ? W[(size_t)kk * Ncols + col] : 0.0f;
    outp[((size_t)(frag * 32 + lane)) * 16 + j] = (__bf16)v;
  }
}

// ===========================================================================
// Kernel 3: fused MLP chain on WMMA.
// Block = 128 threads = 4 waves; each wave owns 16 rows (points) end-to-end.
// All activations live in LDS as row-major bf16; f32 accumulation in WMMA.
// Layers are templates (compile-time kTiles/strides) and force-inlined so
// InferAddressSpaces resolves A-frag loads to ds_load_b128 and B-frag loads
// to global_load_b128 (no generic FLAT path).
// ===========================================================================
__device__ PCMPE_INLINE bf16x16 pcmpe_load_a(const __bf16* base, int stride,
                                             int kt, int lane) {
  // A(16x32 bf16): lane = m + 16h ; elements 0..7 <- K = kt*32 + 8h + j
  //                                 elements 8..15 <- K = kt*32 + 16 + 8h + j
  const int h = lane >> 4, m = lane & 15;
  const __bf16* pr = base + m * stride + kt * 32 + h * 8;
  bf16x8 lo = *(const bf16x8*)(pr);
  bf16x8 hi = *(const bf16x8*)(pr + 16);
  return __builtin_shufflevector(lo, hi, 0, 1, 2, 3, 4, 5, 6, 7,
                                 8, 9, 10, 11, 12, 13, 14, 15);
}

__device__ PCMPE_INLINE float pcmpe_gelu(float x) {
  return 0.5f * x * (1.0f + erff(x * 0.70710678118654752f));
}

template <int KT, int SA, int SO, bool GELU>
__device__ PCMPE_INLINE void pcmpe_layer(const __bf16* aBase,
                                         const __bf16* __restrict__ wPack, // nTiles = 8
                                         const float* __restrict__ bias,
                                         __bf16* outBase, int lane) {
  const int h = lane >> 4, nn = lane & 15;
  const bf16x16* __restrict__ wp = (const bf16x16*)wPack;
#pragma unroll
  for (int nt = 0; nt < 8; ++nt) {
    f32x8 acc = {};
#pragma unroll
    for (int kt = 0; kt < KT; ++kt) {
      bf16x16 a = pcmpe_load_a(aBase, SA, kt, lane);
      bf16x16 bb = wp[(kt * 8 + nt) * 32 + lane];
      acc = __builtin_amdgcn_wmma_f32_16x16x32_bf16(false, a, false, bb,
                                                    (short)0, acc, false, false);
    }
    const float bv = bias[nt * 16 + nn];
#pragma unroll
    for (int r = 0; r < 8; ++r) {
      float v = acc[r] + bv;
      if (GELU) v = pcmpe_gelu(v);
      outBase[(r + 8 * h) * SO + nt * 16 + nn] = (__bf16)v;   // row r+8h, col nt*16+nn
    }
  }
}

__global__ void __launch_bounds__(128)
pcmpe_mlp_kernel(const float* __restrict__ invariants,   // [32768,6] f32 (ws)
                 const float* __restrict__ features,     // [32768,64] f32
                 const __bf16* __restrict__ invW1p, const float* __restrict__ inv_b1,
                 const __bf16* __restrict__ invW2p, const float* __restrict__ inv_b2,
                 const __bf16* __restrict__ featW1p, const float* __restrict__ feat_b1,
                 const __bf16* __restrict__ featW2p, const float* __restrict__ feat_b2,
                 const __bf16* __restrict__ shW1p, const float* __restrict__ sh_b1,
                 const __bf16* __restrict__ shW2p, const float* __restrict__ sh_b2,
                 const __bf16* __restrict__ g0p, const float* __restrict__ g0_b,
                 float* __restrict__ out) {
  __shared__ __bf16 sConcat[64 * 256];   // 32KB: concat buffer [row][256]
  __shared__ __bf16 sB[64 * 128];        // 16KB: inter-layer hidden [row][128]

  const int lane = threadIdx.x & 31;
  const int wave = threadIdx.x >> 5;      // 0..3
  const int r0w  = wave * 16;             // wave's first row in tile
  const int p0   = blockIdx.x * 64 + r0w; // wave's first global point

  __bf16* cRow = sConcat + r0w * 256;
  __bf16* bRow = sB      + r0w * 128;

  // --- stage invariants (K=6, zero-pad to 32) into concat cols 128..159 ---
#pragma unroll
  for (int t = 0; t < 16; ++t) {
    const int idx = t * 32 + lane;          // 16*32 elements, 32 lanes
    const int r = idx >> 5, c = idx & 31;
    const float v = (c < 6) ? invariants[(size_t)(p0 + r) * 6 + c] : 0.0f;
    cRow[r * 256 + 128 + c] = (__bf16)v;
  }
  // inv MLP
  pcmpe_layer<1, 256, 128, true >(cRow + 128, invW1p, inv_b1, bRow, lane);
  pcmpe_layer<4, 128, 256, false>(bRow,       invW2p, inv_b2, cRow, lane);  // concat[:,0:128]

  // --- stage features (K=64) into concat cols 128..191 ---
#pragma unroll
  for (int t = 0; t < 32; ++t) {
    const int idx = t * 32 + lane;          // 16*64 elements
    const int r = idx >> 6, c = idx & 63;
    cRow[r * 256 + 128 + c] = (__bf16)features[(size_t)(p0 + r) * 64 + c];
  }
  // feat MLP
  pcmpe_layer<2, 256, 128, true >(cRow + 128, featW1p, feat_b1, bRow,       lane);
  pcmpe_layer<4, 128, 256, false>(bRow,       featW2p, feat_b2, cRow + 128, lane); // concat[:,128:256]

  // shared MLP over concat (K=256)
  pcmpe_layer<8, 256, 128, true >(cRow, shW1p, sh_b1, bRow, lane);
  pcmpe_layer<4, 128, 256, false>(bRow, shW2p, sh_b2, cRow, lane);  // hidden -> concat[:,0:128]

  // g0 head: scalar = hidden @ g0_W + g0_b (single N-tile, only col 0 valid)
  {
    const bf16x16* __restrict__ wp = (const bf16x16*)g0p;
    f32x8 acc = {};
#pragma unroll
    for (int kt = 0; kt < 4; ++kt) {
      bf16x16 a = pcmpe_load_a(cRow, 256, kt, lane);
      bf16x16 bb = wp[kt * 32 + lane];
      acc = __builtin_amdgcn_wmma_f32_16x16x32_bf16(false, a, false, bb,
                                                    (short)0, acc, false, false);
    }
    const int h = lane >> 4, nn = lane & 15;
    if (nn == 0) {
      const float gb = g0_b[0];
#pragma unroll
      for (int r = 0; r < 8; ++r)
        out[(size_t)(p0 + r + 8 * h) * 16 + 0] = acc[r] + gb;
    }
  }
}

// ===========================================================================
// Host-side launcher
// ===========================================================================
extern "C" void kernel_launch(void* const* d_in, const int* in_sizes, int n_in,
                              void* d_out, int out_size, void* d_ws, size_t ws_size,
                              hipStream_t stream) {
  (void)in_sizes; (void)n_in; (void)out_size; (void)ws_size;
  const float* coords   = (const float*)d_in[0];
  const float* features = (const float*)d_in[1];
  const float* inv_W1   = (const float*)d_in[2];
  const float* inv_b1   = (const float*)d_in[3];
  const float* inv_W2   = (const float*)d_in[4];
  const float* inv_b2   = (const float*)d_in[5];
  const float* feat_W1  = (const float*)d_in[6];
  const float* feat_b1  = (const float*)d_in[7];
  const float* feat_W2  = (const float*)d_in[8];
  const float* feat_b2  = (const float*)d_in[9];
  const float* sh_W1    = (const float*)d_in[10];
  const float* sh_b1    = (const float*)d_in[11];
  const float* sh_W2    = (const float*)d_in[12];
  const float* sh_b2    = (const float*)d_in[13];
  const float* g0_W     = (const float*)d_in[14];
  const float* g0_b     = (const float*)d_in[15];
  float* out = (float*)d_out;

  char* ws = (char*)d_ws;
  float*  wCenter = (float*)(ws + WS_CENTER);
  float*  wInv    = (float*)(ws + WS_INV);
  char*   wts     = ws + WS_WTS;
  __bf16* pInvW1  = (__bf16*)(wts + WP_INV_W1);
  __bf16* pInvW2  = (__bf16*)(wts + WP_INV_W2);
  __bf16* pFeatW1 = (__bf16*)(wts + WP_FEAT_W1);
  __bf16* pFeatW2 = (__bf16*)(wts + WP_FEAT_W2);
  __bf16* pShW1   = (__bf16*)(wts + WP_SH_W1);
  __bf16* pShW2   = (__bf16*)(wts + WP_SH_W2);
  __bf16* pG0     = (__bf16*)(wts + WP_G0);

  // 1) centroid per batch
  pcmpe_center_kernel<<<BATCH, 256, 0, stream>>>(coords, wCenter);

  // 2) kNN + geometry (invariants -> ws, grades 1..4 -> d_out)
  pcmpe_geom_kernel<<<NPOINTS / 256, 256, 0, stream>>>(coords, wCenter, wInv, out);

  // 3) pack weights into WMMA B-fragment bf16 layout
  pcmpe_pack_kernel<<<1, 256, 0, stream>>>(inv_W1,  pInvW1,  6,   128, 1, 8);
  pcmpe_pack_kernel<<<4, 256, 0, stream>>>(inv_W2,  pInvW2,  128, 128, 4, 8);
  pcmpe_pack_kernel<<<2, 256, 0, stream>>>(feat_W1, pFeatW1, 64,  128, 2, 8);
  pcmpe_pack_kernel<<<4, 256, 0, stream>>>(feat_W2, pFeatW2, 128, 128, 4, 8);
  pcmpe_pack_kernel<<<8, 256, 0, stream>>>(sh_W1,   pShW1,   256, 128, 8, 8);
  pcmpe_pack_kernel<<<4, 256, 0, stream>>>(sh_W2,   pShW2,   128, 128, 4, 8);
  pcmpe_pack_kernel<<<1, 256, 0, stream>>>(g0_W,    pG0,     128, 1,   4, 1);

  // 4) fused WMMA MLP chain (scalar -> d_out col 0)
  pcmpe_mlp_kernel<<<NPOINTS / 64, 128, 0, stream>>>(
      wInv, features,
      pInvW1, inv_b1, pInvW2, inv_b2,
      pFeatW1, feat_b1, pFeatW2, feat_b2,
      pShW1, sh_b1, pShW2, sh_b2,
      pG0, g0_b, out);
}
